// Model_39676907887188
// MI455X (gfx1250) — matrix operand order repaired and checked by reading, not compiled, not run on hardware
//
#include <hip/hip_runtime.h>
#include <hip/hip_bf16.h>
#include <cstdint>

typedef _Float16 f16;
typedef __attribute__((ext_vector_type(4)))  _Float16 v4h;
typedef __attribute__((ext_vector_type(16))) _Float16 v16h;
typedef __attribute__((ext_vector_type(8)))  float    v8f;

// Problem constants (reference: B=8, S=2048, D=1024, Dv=1024, scale=0.1)
constexpr int Bn  = 8;
constexpr int Sn  = 2048;
constexpr int Dn  = 1024;
constexpr int DVn = 1024;
constexpr float SCALE = 0.1f;

// Tiling
constexpr int MB = 64;         // query rows per workgroup
constexpr int NB = 64;         // key block
constexpr int KD = 128;        // K-dim staging depth per slab
constexpr int KC = 32;         // WMMA K (f16)
constexpr int NTHREADS = 512;  // 16 wave32
constexpr int NCHUNK = Dn / KD;
constexpr int NKB = Sn / NB;

// Padded LDS strides (KSTR/QSTR give 4-bank lane skew -> conflict-free b128 reads)
constexpr int QSTR = Dn + 8;   // f16 per Q row   (1032)
constexpr int KSTR = KD + 8;   // f16 per K^T row (136)
constexpr int SSTR = NB + 4;   // f32 per S row   (68)
constexpr int PSTR = NB + 8;   // f16 per P row   (72)
constexpr int KSLAB = NB * KSTR;

// LDS byte offsets
constexpr size_t OFF_Q = 0;
constexpr size_t OFF_K = OFF_Q + (size_t)MB * QSTR * sizeof(f16);
constexpr size_t OFF_S = OFF_K + 2 * (size_t)KSLAB * sizeof(f16);
constexpr size_t OFF_P = OFF_S + (size_t)MB * SSTR * sizeof(float);
constexpr size_t OFF_R = OFF_P + (size_t)MB * PSTR * sizeof(f16);
constexpr size_t OFF_M = OFF_R + 8 * MB * sizeof(float);
constexpr size_t OFF_L = OFF_M + MB * sizeof(float);
constexpr size_t OFF_C = OFF_L + MB * sizeof(float);
constexpr size_t SMEM_BYTES = OFF_C + MB * sizeof(float);   // ~192 KB < 320 KB WGP LDS

__global__ __launch_bounds__(NTHREADS, 1)
void flash_attn_wmma(const float* __restrict__ Q,
                     const float* __restrict__ K,
                     const float* __restrict__ V,
                     float* __restrict__ OUT) {
  extern __shared__ char smem[];
  f16*   Qs   = (f16*)(smem + OFF_Q);   // [MB][QSTR]  (m, d)
  f16*   Ks   = (f16*)(smem + OFF_K);   // 2 x [NB][KSTR], transposed (t, d_local)
  float* Sb   = (float*)(smem + OFF_S); // [MB][SSTR]
  f16*   Pb   = (f16*)(smem + OFF_P);   // [MB][PSTR]
  float* Red  = (float*)(smem + OFF_R); // [8][MB] partial max/sum
  float* Mrow = (float*)(smem + OFF_M);
  float* Lrow = (float*)(smem + OFF_L);
  float* Cf   = (float*)(smem + OFF_C);

  const int tid  = threadIdx.x;
  const int lane = tid & 31;
  const int wave = tid >> 5;
  const int ln16 = lane & 15;
  const int hi   = lane >> 4;

  const int b  = blockIdx.y;
  const int m0 = blockIdx.x * MB;

  const float* qp = Q + (size_t)b * Sn * Dn + (size_t)m0 * Dn;
  const float* kp = K + (size_t)b * Dn * Sn;
  const float* vp = V + (size_t)b * Sn * DVn;
  float*       op = OUT + (size_t)b * Sn * DVn + (size_t)m0 * DVn;

  // K-staging geometry: thread owns 4-float t-column group, 4 *consecutive* d-rows
  const int c4x = (tid & 15) * 4;      // t-column offset within key block
  const int dg  = tid >> 4;            // d-row group (0..31) -> rows dg*4 .. dg*4+3

  auto load_chunk = [&](int kd, int n0_, float4 (&r)[4], bool pf) {
    #pragma unroll
    for (int j = 0; j < 4; ++j) {
      const float* src = kp + (size_t)(kd * KD + dg * 4 + j) * Sn + n0_ + c4x;
      r[j] = *reinterpret_cast<const float4*>(src);
      if (pf) __builtin_prefetch(src + (size_t)KD * Sn, 0, 1);
    }
  };
  auto cvt_store_slab = [&](f16* dslab, const float4 (&r)[4]) {
    const v4h w0 = {(f16)r[0].x, (f16)r[1].x, (f16)r[2].x, (f16)r[3].x};
    const v4h w1 = {(f16)r[0].y, (f16)r[1].y, (f16)r[2].y, (f16)r[3].y};
    const v4h w2 = {(f16)r[0].z, (f16)r[1].z, (f16)r[2].z, (f16)r[3].z};
    const v4h w3 = {(f16)r[0].w, (f16)r[1].w, (f16)r[2].w, (f16)r[3].w};
    *reinterpret_cast<v4h*>(dslab + (c4x + 0) * KSTR + dg * 4) = w0;  // [t][d] transposed,
    *reinterpret_cast<v4h*>(dslab + (c4x + 1) * KSTR + dg * 4) = w1;  // 4 f16 contiguous in d
    *reinterpret_cast<v4h*>(dslab + (c4x + 2) * KSTR + dg * 4) = w2;
    *reinterpret_cast<v4h*>(dslab + (c4x + 3) * KSTR + dg * 4) = w3;
  };

  // ---- Stage Q tile (64 x 1024) as f16 into LDS, packed b64 stores ----
  for (int i = tid; i < MB * Dn / 4; i += NTHREADS) {
    const int m  = i >> 8;
    const int c4 = i & 255;
    const float4 f = *reinterpret_cast<const float4*>(qp + (size_t)m * Dn + c4 * 4);
    const v4h w = {(f16)f.x, (f16)f.y, (f16)f.z, (f16)f.w};
    *reinterpret_cast<v4h*>(Qs + m * QSTR + c4 * 4) = w;
  }
  // preload K chunk 0 of key block 0 into slab 0
  {
    float4 r[4];
    load_chunk(0, 0, r, true);
    cvt_store_slab(Ks, r);
  }
  if (tid < MB) { Mrow[tid] = -3.0e38f; Lrow[tid] = 0.0f; }

  v8f acc[16] = {};                    // 64(M) x 64(Dv-slice) f32 per wave

  const int mi   = wave >> 2;
  const int ni   = wave & 3;
  const int tcol = ni * 16 + ln16;
  const int asel = hi ? 16 : 0;
  const int bsel = hi ? 16 : 0;
  const int am   = mi * 16 + ln16;
  const int dvb  = wave * 64;

  __syncthreads();

  for (int kb = 0; kb < NKB; ++kb) {
    const int n0 = kb * NB;

    // ====== Phase A: S = (Q K) * 0.1, double-buffered K slabs ======
    v8f cS0 = {}, cS1 = {};
    for (int kd = 0; kd < NCHUNK; ++kd) {
      const bool more = (kd + 1 < NCHUNK);
      float4 r[4];
      if (more) load_chunk(kd + 1, n0, r, kd + 2 < NCHUNK);
      const f16* slab = Ks + (kd & 1) * KSLAB;
      // Batch all fragment loads of this chunk up front so the DS pipe runs
      // ahead of the WMMA stream (staggered s_wait_dscnt instead of 0).
      v16h aQ[4], bK[4];
      #pragma unroll
      for (int k4 = 0; k4 < 4; ++k4) {
        const f16* qrow = Qs + am * QSTR + kd * KD + k4 * KC + asel;
        #pragma unroll
        for (int e = 0; e < 8; ++e) { aQ[k4][e] = qrow[e]; aQ[k4][8 + e] = qrow[8 + e]; }
        const f16* krow = slab + tcol * KSTR + k4 * KC + bsel;
        #pragma unroll
        for (int e = 0; e < 16; ++e) bK[k4][e] = krow[e];
      }
      #pragma unroll
      for (int k4 = 0; k4 < 4; ++k4) {
        if ((k4 & 1) == 0)
          cS0 = __builtin_amdgcn_wmma_f32_16x16x32_f16(false, aQ[k4], false, bK[k4],
                                                       (short)0, cS0, false, false);
        else
          cS1 = __builtin_amdgcn_wmma_f32_16x16x32_f16(false, aQ[k4], false, bK[k4],
                                                       (short)0, cS1, false, false);
      }
      if (more) {
        cvt_store_slab(Ks + ((kd + 1) & 1) * KSLAB, r);
        __syncthreads();
      }
    }
    #pragma unroll
    for (int jj = 0; jj < 8; ++jj) {
      const int m = mi * 16 + jj + 8 * hi;
      Sb[m * SSTR + tcol] = (cS0[jj] + cS1[jj]) * SCALE;
    }
    __syncthreads();

    // ====== Phase B: online softmax, all 512 threads ======
    const int r8 = tid & 63;
    const int sg = tid >> 6;
    const float* srow = Sb + r8 * SSTR + sg * 8;
    const float4 x = *reinterpret_cast<const float4*>(srow);
    const float4 y = *reinterpret_cast<const float4*>(srow + 4);
    Red[sg * MB + r8] = fmaxf(fmaxf(fmaxf(x.x, x.y), fmaxf(x.z, x.w)),
                              fmaxf(fmaxf(y.x, y.y), fmaxf(y.z, y.w)));
    __syncthreads();
    if (tid < MB) {
      float mold = Mrow[tid], mnew = mold;
      #pragma unroll
      for (int s = 0; s < 8; ++s) mnew = fmaxf(mnew, Red[s * MB + tid]);
      Cf[tid]   = __expf(mold - mnew);
      Mrow[tid] = mnew;
    }
    __syncthreads();
    {
      const float mnew = Mrow[r8];
      float p[8];
      p[0] = __expf(x.x - mnew); p[1] = __expf(x.y - mnew);
      p[2] = __expf(x.z - mnew); p[3] = __expf(x.w - mnew);
      p[4] = __expf(y.x - mnew); p[5] = __expf(y.y - mnew);
      p[6] = __expf(y.z - mnew); p[7] = __expf(y.w - mnew);
      f16* prow = Pb + r8 * PSTR + sg * 8;
      float ps = 0.0f;
      #pragma unroll
      for (int e = 0; e < 8; ++e) { ps += p[e]; prow[e] = (f16)p[e]; }
      Red[sg * MB + r8] = ps;
    }
    __syncthreads();
    if (tid < MB) {   // overlaps with Phase C of the other waves
      float l = Lrow[tid] * Cf[tid];
      #pragma unroll
      for (int s = 0; s < 8; ++s) l += Red[s * MB + tid];
      Lrow[tid] = l;
    }

    // Cross-block pipelining: next key block's K chunk-0 loads ride under the PV WMMAs.
    const bool morekb = (kb + 1 < NKB);
    float4 rn[4];
    if (morekb) load_chunk(0, n0 + NB, rn, true);

    // ====== Phase C: acc = acc*cf + P @ V (per-wave Dv slice) ======
    #pragma unroll
    for (int m2 = 0; m2 < 4; ++m2) {
      float cfv[8];
      const int mbase = m2 * 16 + 8 * hi;
      #pragma unroll
      for (int jj = 0; jj < 8; ++jj) cfv[jj] = Cf[mbase + jj];
      #pragma unroll
      for (int nj = 0; nj < 4; ++nj) {
        v8f& c = acc[m2 * 4 + nj];
        #pragma unroll
        for (int jj = 0; jj < 8; ++jj) c[jj] *= cfv[jj];
      }
    }
    #pragma unroll
    for (int kc = 0; kc < NB; kc += KC) {
      v16h aP[4];
      #pragma unroll
      for (int m2 = 0; m2 < 4; ++m2) {
        const f16* prow = Pb + (m2 * 16 + ln16) * PSTR + kc + asel;
        #pragma unroll
        for (int e = 0; e < 8; ++e) { aP[m2][e] = prow[e]; aP[m2][8 + e] = prow[8 + e]; }
      }
      const float* vbase = vp + (size_t)(n0 + kc + bsel) * DVn;
      #pragma unroll
      for (int nj = 0; nj < 4; ++nj) {
        const int dv = dvb + nj * 16 + ln16;
        v16h bV;
        #pragma unroll
        for (int e = 0; e < 16; ++e) bV[e] = (f16)vbase[(size_t)e * DVn + dv];
        #pragma unroll
        for (int m2 = 0; m2 < 4; ++m2)
          acc[m2 * 4 + nj] = __builtin_amdgcn_wmma_f32_16x16x32_f16(
              false, aP[m2], false, bV, (short)0, acc[m2 * 4 + nj], false, false);
      }
    }
    // No barrier needed here: Phase C touches only Pb/Cf/global V, and the last
    // reader of K slab 0 was fenced by its own chunk barrier.  The tail store
    // below is made visible to next block's kd=0 by the single barrier after it.
    if (morekb) cvt_store_slab(Ks, rn);   // slab 0 for next block
    __syncthreads();
  }

  // ====== Epilogue: divide by softmax denominator, write fp32 ======
  #pragma unroll
  for (int m2 = 0; m2 < 4; ++m2) {
    const int mbase = m2 * 16 + 8 * hi;
    float linv[8];
    #pragma unroll
    for (int jj = 0; jj < 8; ++jj) linv[jj] = 1.0f / Lrow[mbase + jj];
    #pragma unroll
    for (int nj = 0; nj < 4; ++nj) {
      const int dv = dvb + nj * 16 + ln16;
      const v8f c = acc[m2 * 4 + nj];
      #pragma unroll
      for (int jj = 0; jj < 8; ++jj)
        op[(size_t)(mbase + jj) * DVn + dv] = c[jj] * linv[jj];
    }
  }
}

extern "C" void kernel_launch(void* const* d_in, const int* in_sizes, int n_in,
                              void* d_out, int out_size, void* d_ws, size_t ws_size,
                              hipStream_t stream) {
  (void)in_sizes; (void)n_in; (void)d_ws; (void)ws_size; (void)out_size;
  const float* q = (const float*)d_in[0];
  const float* k = (const float*)d_in[1];
  const float* v = (const float*)d_in[2];
  float* out = (float*)d_out;

  static bool attr_set = false;
  if (!attr_set) {
    hipFuncSetAttribute((const void*)flash_attn_wmma,
                        hipFuncAttributeMaxDynamicSharedMemorySize,
                        (int)SMEM_BYTES);
    attr_set = true;
  }

  dim3 grid(Sn / MB, Bn);
  dim3 block(NTHREADS);
  flash_attn_wmma<<<grid, block, SMEM_BYTES, stream>>>(q, k, v, out);
}
